// ScaledDotProductAttention_38062000177659
// MI455X (gfx1250) — matrix operand order repaired and checked by reading, not compiled, not run on hardware
//
#include <hip/hip_runtime.h>
#include <stdint.h>

// Flash attention B=4,H=16,S=2048,D=128, fp32 I/O, bf16 WMMA compute.
// CDNA5 / gfx1250: wave32, v_wmma_f32_16x16x32_bf16.
// K: depth-3 TDM (tensor_load_to_lds) pipeline into fp32 staging; one shared
//    fp32->bf16 conversion per workgroup, overlapped with compute.
// V: register-staged global loads -> swizzled transposed bf16 LDS tile (VT).
// Softmax row-sums computed by an extra WMMA against an all-ones B fragment
// (rowsum(P) = P x ones), removing half of the ds_bpermute reduction traffic.

#define S_LEN 2048
#define D_DIM 128
#define QTILE 128      // q rows per workgroup (8 waves x 16)
#define KTILE 32       // keys per iteration
#define NITER (S_LEN / KTILE)
#define KSTR  132      // fp32 staging row stride (floats), TDM pad 4 dwords
#define KBSTR 136      // bf16 K tile row stride (bf16 units), 16B-aligned rows
#define VTSTR 40       // bf16 VT row stride (bf16 units), rows are d, cols key
#define PSTR  36       // fp32 P scratch row stride (floats)

// VT swizzle (bf16 units, 16B multiple): row bits [6:4] vary per-lane on the
// transpose STORE side (conflict-free banks), constant per READ instruction.
#define VSWZB(row) ((((row) >> 4) & 7) << 3)

typedef __bf16    v16bf __attribute__((ext_vector_type(16)));
typedef __bf16    v8bf  __attribute__((ext_vector_type(8)));
typedef float     v8f   __attribute__((ext_vector_type(8)));
typedef float     f4    __attribute__((ext_vector_type(4)));
typedef uint32_t  u32x4 __attribute__((ext_vector_type(4)));
typedef int       i32x4 __attribute__((ext_vector_type(4)));
typedef int       i32x8 __attribute__((ext_vector_type(8)));

#if __has_builtin(__builtin_amdgcn_tensor_load_to_lds) && __has_builtin(__builtin_amdgcn_s_wait_tensorcnt)
#define USE_TDM 1
#else
#define USE_TDM 0
#endif

__device__ __forceinline__ v8f wmma_bf16(v16bf a, v16bf b, v8f c) {
  // D = A(16x32 bf16) * B(32x16 bf16) + C(16x16 f32)
  return __builtin_amdgcn_wmma_f32_16x16x32_bf16(false, a, false, b, (short)0, c,
                                                 false, false);
}

// 16 contiguous bf16 from LDS -> v16bf (2 x b128, no conversion)
__device__ __forceinline__ v16bf row16_ld(const __bf16* p) {
  v8bf lo = *(const v8bf*)p;
  v8bf hi = *(const v8bf*)(p + 8);
  v16bf r;
#pragma unroll
  for (int e = 0; e < 8; ++e) { r[e] = lo[e]; r[8 + e] = hi[e]; }
  return r;
}

#if USE_TDM
// DMA one KTILE x D_DIM fp32 tile from global into LDS at lds_byte_addr with
// +4 dwords of LDS padding after every 128-dword row -> 132-float row stride.
__device__ __forceinline__ void tdm_load_tile(const float* gsrc, uint32_t lds_byte_addr) {
  uint64_t ga = (uint64_t)(uintptr_t)gsrc;
  u32x4 g0;
  g0[0] = 1u;                                                // count=1 (valid user D#)
  g0[1] = lds_byte_addr;                                     // lds_addr
  g0[2] = (uint32_t)ga;                                      // global_addr[31:0]
  g0[3] = (uint32_t)((ga >> 32) & 0x1FFFFFFu) | (2u << 30);  // global_addr[56:32] | type=2
  i32x8 g1;
  g1[0] = (int)((2u << 16)      // data_size = 4 bytes
              | (1u << 20)      // pad_enable
              | (6u << 22)      // pad_interval: 128 dwords
              | (3u << 25));    // pad_amount: 4 dwords
  g1[1] = (int)(128u << 16);    // tensor_dim0 = 128 (low 16 bits at [31:16])
  g1[2] = (int)(32u  << 16);    // tensor_dim0 hi = 0 | tensor_dim1 = 32
  g1[3] = (int)(128u << 16);    // tensor_dim1 hi = 0 | tile_dim0 = 128
  g1[4] = (int)(32u);           // tile_dim1 = 32 | tile_dim2 = 0
  g1[5] = (int)(128u);          // tensor_dim0_stride = 128 elements
  g1[6] = 0;
  g1[7] = 0;
  i32x4 z4 = {0, 0, 0, 0};
#if defined(__clang_major__) && (__clang_major__ >= 23)
  i32x8 z8 = {0, 0, 0, 0, 0, 0, 0, 0};
  __builtin_amdgcn_tensor_load_to_lds(g0, g1, z4, z4, z8, 0);
#else
  __builtin_amdgcn_tensor_load_to_lds(g0, g1, z4, z4, 0);
#endif
}
#endif

__global__ __launch_bounds__(256) void sdpa_fattn_kernel(
    const float* __restrict__ Q, const float* __restrict__ K,
    const float* __restrict__ V, float* __restrict__ O) {
#if USE_TDM
  __shared__ float ldsKst[2][KTILE * KSTR];      // fp32 staging (TDM target)
#endif
  __shared__ __bf16 ldsKbf[2][KTILE * KBSTR];    // bf16 K tiles (shared convert)
  __shared__ __bf16 ldsVT[D_DIM * VTSTR + 64];   // bf16 transposed V tile
  __shared__ float  ldsP[8 * 16 * PSTR];

  const int tid  = threadIdx.x;
  const int wave = tid >> 5;
  const int lane = tid & 31;
  const int hi   = lane >> 4;   // half-wave: 0/1
  const int ln   = lane & 15;

  const int qt = blockIdx.x;          // q tile (0..15)
  const int bh = blockIdx.y;          // fused batch*head (0..63)
  const size_t base = (size_t)bh * S_LEN * D_DIM;
  const float* Qp = Q + base;
  const float* Kp = K + base;
  const float* Vp = V + base;
  float*       Op = O + base;

  // Per-thread tile segment mapping: 16 contiguous elements of one row.
  const int vkey = tid >> 3;          // row 0..31
  const int vcol = (tid & 7) << 4;    // col 0,16,..,112

  // V staging registers: current tile's 16 floats.
  f4 vreg[4];
  {
    const f4* gv = (const f4*)(Vp + (size_t)vkey * D_DIM + vcol);
#pragma unroll
    for (int j = 0; j < 4; ++j) vreg[j] = gv[j];
  }

#if USE_TDM
  // Prologue: depth-3 TDM pipeline; convert tile 0 to bf16 before the loop.
  if (wave == 0) {
    tdm_load_tile(Kp,                             (uint32_t)(uintptr_t)&ldsKst[0][0]);
    tdm_load_tile(Kp + (size_t)1 * KTILE * D_DIM, (uint32_t)(uintptr_t)&ldsKst[1][0]);
    tdm_load_tile(Kp + (size_t)2 * KTILE * D_DIM, (uint32_t)(uintptr_t)&ldsKst[0][0]);
  }
#endif

  // ---- Q as 4 A-fragments (16x32 bf16 each). A elem e -> d = e+16*hi
  const int qrow = qt * QTILE + wave * 16 + ln;
  const float* qr = Qp + (size_t)qrow * D_DIM;
  v16bf aQ[4];
#pragma unroll
  for (int c = 0; c < 4; ++c) {
    const float* p0 = qr + 32 * c + 16 * hi;
    v16bf a;
#pragma unroll
    for (int j = 0; j < 2; ++j) {
      f4 x = *(const f4*)(p0 + 4 * j);
      f4 y = *(const f4*)(p0 + 8 + 4 * j);
#pragma unroll
      for (int e = 0; e < 4; ++e) {
        a[4 * j + e]     = (__bf16)x[e];
        a[8 + 4 * j + e] = (__bf16)y[e];
      }
    }
    aQ[c] = a;
  }

  // Constant all-ones B fragment: rowsum(P) = P x ones (one extra WMMA).
  v16bf onesB;
#pragma unroll
  for (int e = 0; e < 16; ++e) onesB[e] = (__bf16)1.0f;

#if USE_TDM
  // Wait for tile 0 (two newer loads may be outstanding), publish, convert.
  if (wave == 0) __builtin_amdgcn_s_wait_tensorcnt(2);
  __syncthreads();
  {
    const float* s = &ldsKst[0][vkey * KSTR + vcol];
    __bf16*      d = &ldsKbf[0][vkey * KBSTR + vcol];
    f4 x[4];
#pragma unroll
    for (int j = 0; j < 4; ++j) x[j] = ((const f4*)s)[j];
#pragma unroll
    for (int j = 0; j < 2; ++j) {
      v8bf o;
#pragma unroll
      for (int e = 0; e < 8; ++e) o[e] = (__bf16)x[2 * j + (e >> 2)][e & 3];
      *(v8bf*)(d + 8 * j) = o;
    }
  }
#endif

  // O accumulator (16x128 f32, C layout: vgpr i = row i+8*hi, col = 16*c+ln)
  v8f acc[8];
#pragma unroll
  for (int c = 0; c < 8; ++c)
#pragma unroll
    for (int i = 0; i < 8; ++i) acc[c][i] = 0.0f;

  float mrow[8], lrow[8];
#pragma unroll
  for (int i = 0; i < 8; ++i) { mrow[i] = -INFINITY; lrow[i] = 0.0f; }

  float* pscr = ldsP + wave * 16 * PSTR;
  const float LOG2E = 1.44269504088896340736f;

  for (int kt = 0; kt < NITER; ++kt) {
    const int buf = kt & 1;
#if USE_TDM
    // In-order TDM: <=1 outstanding => staging tile kt+1 has landed.
    if (wave == 0) __builtin_amdgcn_s_wait_tensorcnt(1);
#endif
    // ---- transpose-store this tile's V registers into VT as bf16
#pragma unroll
    for (int j = 0; j < 16; ++j) {
      const int row = vcol + j;  // d index
      ldsVT[row * VTSTR + VSWZB(row) + vkey] = (__bf16)vreg[j >> 2][j & 3];
    }
#if !USE_TDM
    {  // fallback: K tile kt straight from global, converted to bf16
      const float* gk = Kp + ((size_t)kt * KTILE + vkey) * D_DIM + vcol;
      __bf16*      d  = &ldsKbf[buf][vkey * KBSTR + vcol];
      f4 x[4];
#pragma unroll
      for (int j = 0; j < 4; ++j) x[j] = ((const f4*)gk)[j];
#pragma unroll
      for (int j = 0; j < 2; ++j) {
        v8bf o;
#pragma unroll
        for (int e = 0; e < 8; ++e) o[e] = (__bf16)x[2 * j + (e >> 2)][e & 3];
        *(v8bf*)(d + 8 * j) = o;
      }
    }
#endif
    __syncthreads();  // VT(kt), Kbf[buf](kt), staging(kt+1) all published

    // ---- prefetch next V tile into registers (hidden under compute)
    if (kt + 1 < NITER) {
      const f4* gv = (const f4*)(Vp + ((size_t)(kt + 1) * KTILE + vkey) * D_DIM + vcol);
#pragma unroll
      for (int j = 0; j < 4; ++j) vreg[j] = gv[j];
    }

#if USE_TDM
    // ---- convert staging(kt+1) -> Kbf[buf^1], overlapped with compute below
    if (kt + 1 < NITER) {
      const float* s = &ldsKst[(kt + 1) & 1][vkey * KSTR + vcol];
      __bf16*      d = &ldsKbf[buf ^ 1][vkey * KBSTR + vcol];
      f4 x[4];
#pragma unroll
      for (int j = 0; j < 4; ++j) x[j] = ((const f4*)s)[j];
#pragma unroll
      for (int j = 0; j < 2; ++j) {
        v8bf o;
#pragma unroll
        for (int e = 0; e < 8; ++e) o[e] = (__bf16)x[2 * j + (e >> 2)][e & 3];
        *(v8bf*)(d + 8 * j) = o;
      }
    }
#endif

    const __bf16* kb = &ldsKbf[buf][0];

    // ---- scores: two 16x16 tiles (keys kb+ln and kb+16+ln), 4 d-chunks of K=32
    v8f sc0, sc1;
#pragma unroll
    for (int i = 0; i < 8; ++i) { sc0[i] = 0.0f; sc1[i] = 0.0f; }
#pragma unroll
    for (int c = 0; c < 4; ++c) {
      // B frag: elem e -> B[d = 32c + 16*hi + e][n = ln] = K[key][d]
      v16bf b0 = row16_ld(kb + (ln)      * KBSTR + 32 * c + 16 * hi);
      v16bf b1 = row16_ld(kb + (16 + ln) * KBSTR + 32 * c + 16 * hi);
      sc0 = wmma_bf16(aQ[c], b0, sc0);
      sc1 = wmma_bf16(aQ[c], b1, sc1);
    }

    // ---- online softmax max phase (rows = i + 8*hi, cols across 16-lane halves)
    float rmax[8], corr[8];
#pragma unroll
    for (int i = 0; i < 8; ++i) rmax[i] = fmaxf(sc0[i], sc1[i]);
#pragma unroll
    for (int m = 1; m <= 8; m <<= 1)
#pragma unroll
      for (int i = 0; i < 8; ++i)
        rmax[i] = fmaxf(rmax[i], __shfl_xor(rmax[i], m, 32));
#pragma unroll
    for (int i = 0; i < 8; ++i) {
      float mn = fmaxf(mrow[i], rmax[i]);
      corr[i] = exp2f((mrow[i] - mn) * LOG2E);
      mrow[i] = mn;
      sc0[i] = exp2f((sc0[i] - mn) * LOG2E);
      sc1[i] = exp2f((sc1[i] - mn) * LOG2E);
    }
#pragma unroll
    for (int c = 0; c < 8; ++c)
#pragma unroll
      for (int i = 0; i < 8; ++i) acc[c][i] *= corr[i];

    // ---- P (C layout) -> per-wave LDS scratch -> A-fragment (16x32)
#pragma unroll
    for (int i = 0; i < 8; ++i) {
      pscr[(i + 8 * hi) * PSTR + ln]      = sc0[i];
      pscr[(i + 8 * hi) * PSTR + 16 + ln] = sc1[i];
    }
    v16bf aP;
    {
      const float* pr = pscr + ln * PSTR + 16 * hi;
#pragma unroll
      for (int j = 0; j < 2; ++j) {
        f4 x = *(const f4*)(pr + 4 * j);
        f4 y = *(const f4*)(pr + 8 + 4 * j);
#pragma unroll
        for (int e = 0; e < 4; ++e) {
          aP[4 * j + e]     = (__bf16)x[e];
          aP[8 + 4 * j + e] = (__bf16)y[e];
        }
      }
    }

    // ---- row sums via WMMA: rs = P x ones (all columns equal the row sum)
    {
      v8f z;
#pragma unroll
      for (int i = 0; i < 8; ++i) z[i] = 0.0f;
      v8f rs = wmma_bf16(aP, onesB, z);
#pragma unroll
      for (int i = 0; i < 8; ++i) lrow[i] = lrow[i] * corr[i] + rs[i];
    }

    // ---- O += P(16x32) * V(32x16 per d-chunk); VT rows contiguous bf16 b128
#pragma unroll
    for (int c = 0; c < 8; ++c) {
      const int row = 16 * c + ln;  // d index
      // B frag: elem e -> V[key = 16*hi + e][d = row] = VT[row][16*hi + e]
      v16bf b = row16_ld(ldsVT + row * VTSTR + VSWZB(row) + 16 * hi);
      acc[c] = wmma_bf16(aP, b, acc[c]);
    }

    __syncthreads();  // VT, Kbf[buf], staging(kt+1) free for reuse
#if USE_TDM
    if (wave == 0 && (kt + 3) < NITER) {
      tdm_load_tile(Kp + (size_t)(kt + 3) * KTILE * D_DIM,
                    (uint32_t)(uintptr_t)&ldsKst[(kt + 3) & 1][0]);
    }
#endif
  }

  // ---- epilogue: normalize by l and store fp32
#pragma unroll
  for (int i = 0; i < 8; ++i) lrow[i] = 1.0f / lrow[i];
  const int orow0 = qt * QTILE + wave * 16;
#pragma unroll
  for (int c = 0; c < 8; ++c)
#pragma unroll
    for (int i = 0; i < 8; ++i)
      Op[(size_t)(orow0 + i + 8 * hi) * D_DIM + 16 * c + ln] = acc[c][i] * lrow[i];
}

extern "C" void kernel_launch(void* const* d_in, const int* in_sizes, int n_in,
                              void* d_out, int out_size, void* d_ws, size_t ws_size,
                              hipStream_t stream) {
  (void)in_sizes; (void)n_in; (void)out_size; (void)d_ws; (void)ws_size;
  const float* q = (const float*)d_in[0];
  const float* k = (const float*)d_in[1];
  const float* v = (const float*)d_in[2];
  float* o = (float*)d_out;
  dim3 grid(S_LEN / QTILE, 4 * 16);  // 16 q-tiles x (B*H)=64
  dim3 block(256);                   // 8 waves (wave32)
  sdpa_fattn_kernel<<<grid, block, 0, stream>>>(q, k, v, o);
}
